// Transformer_49529562858425
// MI455X (gfx1250) — compile-verified, matrix operand order checked by reading, not run on hardware
//
#include <hip/hip_runtime.h>
#include <hip/hip_bf16.h>

// ---------------------------------------------------------------------------
// Transformer layer for MI455X (gfx1250): bf16 WMMA + async-to-LDS + TDM.
// B=32, T=512, D=1024, H=16, dh=64, F=4096, M=B*T=16384
// ---------------------------------------------------------------------------

typedef __attribute__((ext_vector_type(16))) __bf16 v16bf;
typedef __attribute__((ext_vector_type(8)))  float  v8f;
typedef __attribute__((ext_vector_type(4)))  unsigned v4u;
typedef __attribute__((ext_vector_type(8)))  unsigned v8u;
typedef unsigned short u16;

#define BB   32
#define TT   512
#define DD   1024
#define HH   16
#define DH   64
#define FF   4096
#define MM   (BB * TT)          // 16384
#define NEG_BIG (-4294967295.0f)

union FragBF {                   // 16 bf16 = 32 bytes = two 16B LDS loads
    v16bf v;
    uint4 q[2];
};

static __device__ __forceinline__ u16 f2bf(float f) {
    union { float f; unsigned u; } x; x.f = f;
    unsigned u = x.u;
    u += 0x7FFFu + ((u >> 16) & 1u);      // round-to-nearest-even
    return (u16)(u >> 16);
}

static __device__ __forceinline__ unsigned lds_off(const void* p) {
    // low 32 bits of the flat address of a __shared__ object == LDS byte offset
    return (unsigned)(unsigned long long)(uintptr_t)p;
}

static __device__ __forceinline__ void async_b128(unsigned lds, const void* gptr) {
    asm volatile("global_load_async_to_lds_b128 %0, %1, off"
                 :: "v"(lds), "v"(gptr) : "memory");
}
static __device__ __forceinline__ void wait_async0() {
    asm volatile("s_wait_asynccnt 0x0" ::: "memory");
}

// ---------------------------------------------------------------------------
// fp32 -> bf16 conversion (plain)
// ---------------------------------------------------------------------------
__global__ __launch_bounds__(256) void cvt_bf16_kernel(const float* __restrict__ src,
                                                       u16* __restrict__ dst, int n) {
    int i = blockIdx.x * 256 + threadIdx.x;
    if (i < n) dst[i] = f2bf(src[i]);
}

// fp32 [K][N] -> bf16 transposed [N][K]
__global__ __launch_bounds__(256) void cvtT_bf16_kernel(const float* __restrict__ src,
                                                        u16* __restrict__ dst,
                                                        int K, int N) {
    int idx = blockIdx.x * 256 + threadIdx.x;
    if (idx < K * N) {
        int k = idx / N, n = idx - k * N;
        dst[(size_t)n * K + k] = f2bf(src[idx]);
    }
}

// ---------------------------------------------------------------------------
// Positional encoding add: q_pe (fp32 + bf16) and k_pe = queries + 2*pe (bf16)
// ---------------------------------------------------------------------------
__global__ __launch_bounds__(256) void pe_add_kernel(const float* __restrict__ q,
                                                     float* __restrict__ qpef,
                                                     u16* __restrict__ qpeb,
                                                     u16* __restrict__ kpeb) {
    int idx = blockIdx.x * 256 + threadIdx.x;           // M*D threads
    int c = idx & (DD - 1);
    int t = (idx >> 10) & (TT - 1);
    float inv   = __expf((float)c * -0.01798894478f);   // -2*ln(10000)/1024
    float angle = (float)t * inv;
    float pe = (((c & 1) == 0) ? __sinf(angle) : __cosf(angle)) * 32.0f;
    float v = q[idx] + pe;
    qpef[idx] = v;
    qpeb[idx] = f2bf(v);
    kpeb[idx] = f2bf(v + pe);
}

// ---------------------------------------------------------------------------
// Generic bf16 GEMM: C[M,N] = A[M,K] * W[K,N], weights pre-transposed Wt[N][K].
// 128x128 block tile, 256 threads (8 waves), each wave 64x32 output.
// A tile staged via global_load_async_to_lds_b128 (ASYNCcnt),
// Wt tile staged via TDM tensor_load_to_lds issued by wave 0 (TENSORcnt).
// mode 0: outB = bf16(C) ; mode 1: outB = bf16(relu C) ;
// mode 2: outF = C + resid (fp32, in place)
// ---------------------------------------------------------------------------
__global__ __launch_bounds__(256) void gemm_bf16_kernel(
        const u16* __restrict__ A, const u16* __restrict__ Wt,
        int M, int N, int K, int mode,
        const float* __restrict__ resid, float* __restrict__ outF,
        u16* __restrict__ outB) {
    __shared__ u16 As[128 * 32];       // [m][k]
    __shared__ u16 Bs[128 * 32];       // [n][k]

    const int tid  = threadIdx.x;
    const int wid  = tid >> 5;
    const int lane = tid & 31;
    const int l16  = lane & 15;
    const int half = lane >> 4;
    const int m0 = blockIdx.y * 128;
    const int n0 = blockIdx.x * 128;
    const int waveM = (wid & 1) * 64;
    const int waveN = (wid >> 1) * 32;
    const unsigned ldsA = lds_off(&As[0]);
    const unsigned ldsB = lds_off(&Bs[0]);

    v8f acc[4][2];
    for (int a = 0; a < 4; ++a)
        for (int b = 0; b < 2; ++b)
            for (int e = 0; e < 8; ++e) acc[a][b][e] = 0.0f;

    for (int kt = 0; kt < K; kt += 32) {
        // ---- A tile: 128x32 bf16 = 8KB, async direct-to-LDS, 16B/lane/op ----
        #pragma unroll
        for (int i = 0; i < 2; ++i) {
            int cidx = i * 256 + tid;                 // 512 chunks of 8 bf16
            int r = cidx >> 2, cc = (cidx & 3) * 8;
            async_b128(ldsA + cidx * 16, &A[(size_t)(m0 + r) * K + kt + cc]);
        }
        // ---- Wt tile: 128 rows x 32 k, one TDM descriptor from wave 0 ----
        if (wid == 0) {
            unsigned long long ga =
                (unsigned long long)(uintptr_t)(Wt + (size_t)n0 * K + kt);
            v4u g0;
            g0[0] = 1u;                               // count=1, user mode
            g0[1] = ldsB;                             // lds_addr
            g0[2] = (unsigned)ga;                     // global_addr[31:0]
            g0[3] = (unsigned)(ga >> 32) | (2u << 30);// global_addr[56:32] | type=2
            v8u g1;
            g1[0] = 0x00010000u;                      // data_size=1 (2 bytes)
            g1[1] = ((unsigned)K & 0xFFFFu) << 16;            // tensor_dim0 lo
            g1[2] = (((unsigned)K >> 16) & 0xFFFFu)           // tensor_dim0 hi
                  | (((unsigned)N & 0xFFFFu) << 16);          // tensor_dim1 lo
            g1[3] = (((unsigned)N >> 16) & 0xFFFFu)           // tensor_dim1 hi
                  | (32u << 16);                              // tile_dim0 = 32
            g1[4] = 128u;                                     // tile_dim1 = 128
            g1[5] = (unsigned)K;                              // dim0 stride lo
            g1[6] = 0u;                                       // stride hi / dim1 stride
            g1[7] = 0u;
            asm volatile("tensor_load_to_lds %0, %1" :: "s"(g0), "s"(g1) : "memory");
        }
        if (kt + 32 < K) {   // hint next A tile (global_prefetch_b8)
            __builtin_prefetch(&A[(size_t)(m0 + (tid >> 1)) * K + kt + 32], 0, 0);
        }
        wait_async0();
        __builtin_amdgcn_s_wait_tensorcnt(0);
        __syncthreads();

        FragBF afr[4], bfr[2];
        #pragma unroll
        for (int tm = 0; tm < 4; ++tm) {
            const u16* p = &As[(waveM + tm * 16 + l16) * 32 + half * 8];
            afr[tm].q[0] = *(const uint4*)p;
            afr[tm].q[1] = *(const uint4*)(p + 16);
        }
        #pragma unroll
        for (int tn = 0; tn < 2; ++tn) {
            const u16* p = &Bs[(waveN + tn * 16 + l16) * 32 + half * 8];
            bfr[tn].q[0] = *(const uint4*)p;
            bfr[tn].q[1] = *(const uint4*)(p + 16);
        }
        #pragma unroll
        for (int tm = 0; tm < 4; ++tm)
            #pragma unroll
            for (int tn = 0; tn < 2; ++tn)
                acc[tm][tn] = __builtin_amdgcn_wmma_f32_16x16x32_bf16(
                    false, afr[tm].v, false, bfr[tn].v,
                    (short)0, acc[tm][tn], false, false);
        __syncthreads();
    }

    // epilogue: C/D layout -> n = l16, m = v + half*8
    #pragma unroll
    for (int tm = 0; tm < 4; ++tm)
        #pragma unroll
        for (int tn = 0; tn < 2; ++tn)
            #pragma unroll
            for (int v = 0; v < 8; ++v) {
                size_t row = (size_t)(m0 + waveM + tm * 16 + v + half * 8);
                size_t col = (size_t)(n0 + waveN + tn * 16 + l16);
                float c = acc[tm][tn][v];
                if (mode == 0) {
                    outB[row * N + col] = f2bf(c);
                } else if (mode == 1) {
                    outB[row * N + col] = f2bf(c > 0.0f ? c : 0.0f);
                } else {
                    outF[row * N + col] = c + resid[row * N + col];
                }
            }
}

// ---------------------------------------------------------------------------
// Attention: one workgroup = (batch b, head h, 64-query tile).
// 64x512 fp32 score strip + 64KB bf16 P strip in LDS (320KB WGP LDS),
// Q/K tiles staged with async-to-LDS b128; fused softmax + masks + residual.
// ---------------------------------------------------------------------------
__global__ __launch_bounds__(256) void attn_kernel(
        const u16* __restrict__ qs, const u16* __restrict__ ks,
        const u16* __restrict__ vs,
        const int* __restrict__ qmask, const int* __restrict__ kmask,
        const float* __restrict__ qpef,
        float* __restrict__ resF, u16* __restrict__ resB) {
    __shared__ float sc[64 * 512];     // 128 KB score strip
    __shared__ u16   pb[64 * 512];     //  64 KB bf16 probabilities
    __shared__ u16   qT[64 * 64];      // Q tile   [q][d]
    __shared__ u16   kv[64 * 64];      // K tile [key][d]  /  V^T tile [d][key]
    __shared__ float red[64 * 4];

    const int tid  = threadIdx.x;
    const int wid  = tid >> 5;
    const int lane = tid & 31;
    const int l16  = lane & 15;
    const int half = lane >> 4;
    const int qt = blockIdx.x, h = blockIdx.y, b = blockIdx.z;
    const int q0 = qt * 64;
    const size_t rowBase = (size_t)b * TT + q0;
    const int colBase = h * DH;
    const int tmS  = wid & 3;          // wave's 16-row tile
    const int tn0  = (wid >> 2) * 2;   // wave's two 16-col tiles
    const unsigned ldsQ = lds_off(&qT[0]);
    const unsigned ldsKV = lds_off(&kv[0]);

    #pragma unroll
    for (int i = 0; i < 2; ++i) {      // Q tile: 64x64 = 512 chunks of 8
        int cidx = i * 256 + tid;
        int r = cidx >> 3, cc = (cidx & 7) * 8;
        async_b128(ldsQ + cidx * 16, &qs[(rowBase + r) * DD + colBase + cc]);
    }
    wait_async0();
    __syncthreads();

    // ---- S = Q K^T, 8 blocks of 64 keys ----
    for (int kb = 0; kb < 8; ++kb) {
        #pragma unroll
        for (int i = 0; i < 2; ++i) {
            int cidx = i * 256 + tid;
            int r = cidx >> 3, cc = (cidx & 7) * 8;
            async_b128(ldsKV + cidx * 16,
                       &ks[((size_t)b * TT + kb * 64 + r) * DD + colBase + cc]);
        }
        wait_async0();
        __syncthreads();
        #pragma unroll
        for (int t = 0; t < 2; ++t) {
            int tn = tn0 + t;
            v8f c;
            for (int e = 0; e < 8; ++e) c[e] = 0.0f;
            #pragma unroll
            for (int kk = 0; kk < 2; ++kk) {
                FragBF a, bb;
                const u16* pa = &qT[(tmS * 16 + l16) * 64 + kk * 32 + half * 8];
                a.q[0] = *(const uint4*)pa;  a.q[1] = *(const uint4*)(pa + 16);
                const u16* pv = &kv[(tn * 16 + l16) * 64 + kk * 32 + half * 8];
                bb.q[0] = *(const uint4*)pv; bb.q[1] = *(const uint4*)(pv + 16);
                c = __builtin_amdgcn_wmma_f32_16x16x32_bf16(
                        false, a.v, false, bb.v, (short)0, c, false, false);
            }
            #pragma unroll
            for (int v = 0; v < 8; ++v)
                sc[(tmS * 16 + v + half * 8) * 512 + kb * 64 + tn * 16 + l16] = c[v];
        }
        __syncthreads();
    }

    // ---- softmax with key mask, then * query mask; 4 threads / row ----
    {
        const int row = tid >> 2, seg = tid & 3;
        const int km = kmask[b], qm = qmask[b];
        float* rp = &sc[row * 512 + seg * 128];
        float mx = -3.0e38f;
        #pragma unroll 4
        for (int j = 0; j < 128; ++j) {
            int col = seg * 128 + j;
            float v = rp[j] * 0.125f;                  // 1/sqrt(64)
            v = (col < km) ? v : NEG_BIG;
            rp[j] = v;
            mx = fmaxf(mx, v);
        }
        red[row * 4 + seg] = mx;
        __syncthreads();
        mx = fmaxf(fmaxf(red[row * 4 + 0], red[row * 4 + 1]),
                   fmaxf(red[row * 4 + 2], red[row * 4 + 3]));
        __syncthreads();
        float sum = 0.0f;
        #pragma unroll 4
        for (int j = 0; j < 128; ++j) {
            float e = __expf(rp[j] - mx);
            rp[j] = e;
            sum += e;
        }
        red[row * 4 + seg] = sum;
        __syncthreads();
        sum = red[row * 4 + 0] + red[row * 4 + 1] + red[row * 4 + 2] + red[row * 4 + 3];
        float scale = ((q0 + row) < qm) ? (1.0f / sum) : 0.0f;
        #pragma unroll 4
        for (int j = 0; j < 128; ++j)
            pb[row * 512 + seg * 128 + j] = f2bf(rp[j] * scale);
    }
    __syncthreads();

    // ---- O = P V, accumulate across 8 key blocks ----
    v8f acc[2];
    for (int t = 0; t < 2; ++t)
        for (int e = 0; e < 8; ++e) acc[t][e] = 0.0f;

    for (int kb = 0; kb < 8; ++kb) {
        #pragma unroll
        for (int i = 0; i < 2; ++i) {   // V^T tile: kv[d][key], vector load + scatter
            int cidx = i * 256 + tid;
            int r = cidx >> 3, cc = (cidx & 7) * 8;
            uint4 dv = *(const uint4*)&vs[((size_t)b * TT + kb * 64 + r) * DD + colBase + cc];
            const u16* e8 = (const u16*)&dv;
            #pragma unroll
            for (int j = 0; j < 8; ++j)
                kv[(cc + j) * 64 + r] = e8[j];
        }
        __syncthreads();
        #pragma unroll
        for (int kk = 0; kk < 2; ++kk) {
            FragBF a;
            const u16* pa = &pb[(tmS * 16 + l16) * 512 + kb * 64 + kk * 32 + half * 8];
            a.q[0] = *(const uint4*)pa; a.q[1] = *(const uint4*)(pa + 16);
            #pragma unroll
            for (int t = 0; t < 2; ++t) {
                FragBF bb;
                const u16* pv = &kv[((tn0 + t) * 16 + l16) * 64 + kk * 32 + half * 8];
                bb.q[0] = *(const uint4*)pv; bb.q[1] = *(const uint4*)(pv + 16);
                acc[t] = __builtin_amdgcn_wmma_f32_16x16x32_bf16(
                            false, a.v, false, bb.v, (short)0, acc[t], false, false);
            }
        }
        __syncthreads();
    }

    // ---- epilogue: + q_pe residual, write fp32 + bf16 ----
    #pragma unroll
    for (int t = 0; t < 2; ++t)
        #pragma unroll
        for (int v = 0; v < 8; ++v) {
            size_t rg = rowBase + tmS * 16 + v + half * 8;
            size_t cg = (size_t)(colBase + (tn0 + t) * 16 + l16);
            float o = acc[t][v] + qpef[rg * DD + cg];
            resF[rg * DD + cg] = o;
            resB[rg * DD + cg] = f2bf(o);
        }
}

// ---------------------------------------------------------------------------
// mean over T: out[b,0,c] = mean_t final[b,t,c]
// ---------------------------------------------------------------------------
__global__ __launch_bounds__(256) void mean_kernel(const float* __restrict__ f,
                                                   float* __restrict__ out) {
    int idx = blockIdx.x * 256 + threadIdx.x;   // 32768
    int b = idx >> 10, c = idx & (DD - 1);
    const float* p = f + (size_t)b * TT * DD + c;
    float s = 0.0f;
    for (int t = 0; t < TT; ++t) s += p[(size_t)t * DD];
    out[idx] = s * (1.0f / (float)TT);
}

// ---------------------------------------------------------------------------
extern "C" void kernel_launch(void* const* d_in, const int* in_sizes, int n_in,
                              void* d_out, int out_size, void* d_ws, size_t ws_size,
                              hipStream_t stream) {
    const float* queries = (const float*)d_in[0];
    // d_in[1] (keys) is unused by the reference computation
    const int*   qmask   = (const int*)d_in[2];
    const int*   kmask   = (const int*)d_in[3];
    const float* Wq      = (const float*)d_in[4];
    const float* Wk      = (const float*)d_in[5];
    const float* Wv      = (const float*)d_in[6];
    const float* fw1     = (const float*)d_in[7];
    const float* fw2     = (const float*)d_in[8];
    float* out = (float*)d_out;

    char* w = (char*)d_ws;
    size_t off = 0;
    auto take = [&](size_t bytes) { char* p = w + off; off += (bytes + 255) & ~(size_t)255; return p; };

    const size_t MD = (size_t)MM * DD;
    float* q_pe_f   = (float*)take(MD * 4);
    float* result_f = (float*)take(MD * 4);
    u16*   q_pe_b   = (u16*)take(MD * 2);
    u16*   k_pe_b   = (u16*)take(MD * 2);
    u16*   qs_b     = (u16*)take(MD * 2);
    u16*   ks_b     = (u16*)take(MD * 2);
    u16*   vs_b     = (u16*)take(MD * 2);
    u16*   result_b = (u16*)take(MD * 2);
    u16*   hmid_b   = (u16*)take((size_t)MM * FF * 2);
    u16*   wqT_b    = (u16*)take((size_t)DD * DD * 2);   // [N][K] transposed
    u16*   wkT_b    = (u16*)take((size_t)DD * DD * 2);
    u16*   wvT_b    = (u16*)take((size_t)DD * DD * 2);
    u16*   fw1T_b   = (u16*)take((size_t)DD * FF * 2);   // [4096][1024]
    u16*   fw2T_b   = (u16*)take((size_t)FF * DD * 2);   // [1024][4096]

    // 1) weights -> bf16, transposed to [N][K]
    int nw = DD * DD;
    cvtT_bf16_kernel<<<(nw + 255) / 256, 256, 0, stream>>>(Wq, wqT_b, DD, DD);
    cvtT_bf16_kernel<<<(nw + 255) / 256, 256, 0, stream>>>(Wk, wkT_b, DD, DD);
    cvtT_bf16_kernel<<<(nw + 255) / 256, 256, 0, stream>>>(Wv, wvT_b, DD, DD);
    int nf = DD * FF;
    cvtT_bf16_kernel<<<(nf + 255) / 256, 256, 0, stream>>>(fw1, fw1T_b, DD, FF);
    cvtT_bf16_kernel<<<(nf + 255) / 256, 256, 0, stream>>>(fw2, fw2T_b, FF, DD);

    // 2) positional encoding
    pe_add_kernel<<<(unsigned)(MD / 256), 256, 0, stream>>>(queries, q_pe_f, q_pe_b, k_pe_b);

    // 3) projections: qs, ks, vs (vs from projected keys, per reference)
    dim3 gProj(DD / 128, MM / 128);
    gemm_bf16_kernel<<<gProj, 256, 0, stream>>>(q_pe_b, wqT_b, MM, DD, DD, 0, nullptr, nullptr, qs_b);
    gemm_bf16_kernel<<<gProj, 256, 0, stream>>>(k_pe_b, wkT_b, MM, DD, DD, 0, nullptr, nullptr, ks_b);
    gemm_bf16_kernel<<<gProj, 256, 0, stream>>>(ks_b,   wvT_b, MM, DD, DD, 0, nullptr, nullptr, vs_b);

    // 4) attention + residual
    dim3 gAttn(TT / 64, HH, BB);
    attn_kernel<<<gAttn, 256, 0, stream>>>(qs_b, ks_b, vs_b, qmask, kmask,
                                           q_pe_f, result_f, result_b);

    // 5) FFN
    dim3 gF1(FF / 128, MM / 128);
    gemm_bf16_kernel<<<gF1, 256, 0, stream>>>(result_b, fw1T_b, MM, FF, DD, 1, nullptr, nullptr, hmid_b);
    dim3 gF2(DD / 128, MM / 128);
    gemm_bf16_kernel<<<gF2, 256, 0, stream>>>(hmid_b, fw2T_b, MM, DD, FF, 2, result_f, result_f, nullptr);

    // 6) mean over T
    mean_kernel<<<(BB * DD) / 256, 256, 0, stream>>>(result_f, out);
}
